// HetGTCN_76682346102815
// MI455X (gfx1250) — matrix operand order repaired and checked by reading, not compiled, not run on hardware
//
#include <hip/hip_runtime.h>
#include <hip/hip_bf16.h>

// ---------------------------------------------------------------------------
// CDNA5 WMMA fragment types / layouts (ISA 7.12.2, wave32):
//   A 16x32 (MxK):  lane L -> row m = L&15, half = L>>4;
//                   elems e hold K = 16*half+e (contiguous, as for B)
//   B 32x16 (KxN):  lane L -> col n = L&15, elems e hold K = 16*(L>>4)+e (contiguous; not ISA table order)
//   C/D 16x16 f32:  lane L -> col n = L&15, VGPR v -> row m = v + 8*(L>>4)
// ---------------------------------------------------------------------------
typedef __attribute__((ext_vector_type(16))) __bf16 v16bf;
typedef __attribute__((ext_vector_type(8)))  __bf16 v8bf;
typedef __attribute__((ext_vector_type(8)))  float  v8f;
typedef __attribute__((ext_vector_type(4)))  unsigned int v4u;
typedef __attribute__((ext_vector_type(8)))  int    v8i;
typedef __attribute__((ext_vector_type(4)))  int    v4i;

// LDS row stride for staged weights: 128 halves + 4-DWORD TDM pad = 136 halves
// (272B -> bank stride 68 dwords == 4 mod 64: conflict-free across 16 lanes)
#define SW_STRIDE 136
#define SW_HALFS  (128 * SW_STRIDE)   // 17408 halves = 128*136*2 = 34816 B

static __device__ __forceinline__ v16bf load_frag_A(const float* __restrict__ A,
                                                    int lda, int m0, int k0) {
  const int lane = threadIdx.x & 31;
  const int m    = m0 + (lane & 15);
  const int half = lane >> 4;
  const float* row = A + (size_t)m * (size_t)lda + k0 + half * 16;
  const float4 lo0 = *(const float4*)(row + 0);
  const float4 lo1 = *(const float4*)(row + 4);
  const float4 hi0 = *(const float4*)(row + 8);
  const float4 hi1 = *(const float4*)(row + 12);
  v16bf a;
  a[0]  = (__bf16)lo0.x; a[1]  = (__bf16)lo0.y; a[2]  = (__bf16)lo0.z; a[3]  = (__bf16)lo0.w;
  a[4]  = (__bf16)lo1.x; a[5]  = (__bf16)lo1.y; a[6]  = (__bf16)lo1.z; a[7]  = (__bf16)lo1.w;
  a[8]  = (__bf16)hi0.x; a[9]  = (__bf16)hi0.y; a[10] = (__bf16)hi0.z; a[11] = (__bf16)hi0.w;
  a[12] = (__bf16)hi1.x; a[13] = (__bf16)hi1.y; a[14] = (__bf16)hi1.z; a[15] = (__bf16)hi1.w;
  return a;
}

// B fragment from LDS-staged bf16 W^T (row n at stride SW_STRIDE halves)
static __device__ __forceinline__ v16bf load_frag_B_lds(const __bf16* sW,
                                                        int kk, int n0) {
  const int lane = threadIdx.x & 31;
  const int n    = n0 + (lane & 15);
  const int kb   = kk + (lane >> 4) * 16;
  const __bf16* p = sW + n * SW_STRIDE + kb;   // 16B-aligned
  v8bf blo = *(const v8bf*)(p);
  v8bf bhi = *(const v8bf*)(p + 8);
  return __builtin_shufflevector(blo, bhi, 0, 1, 2, 3, 4, 5, 6, 7,
                                 8, 9, 10, 11, 12, 13, 14, 15);
}

// ---------------------------------------------------------------------------
// Tensor Data Mover: stage a [nrows x 128] bf16 tile (source row stride = K
// elements) from global W^T into LDS offset 0, with 4-DWORD padding every
// 64 DWORDs (-> SW_STRIDE). D# packing per CDNA5 ISA ch.8.
// ---------------------------------------------------------------------------
static __device__ __forceinline__ void tdm_stage_wt(const __bf16* gsrc,
                                                    int K, int nrows) {
  const unsigned long long ga = (unsigned long long)(size_t)gsrc;
  v4u g0;
  g0[0] = 1u;                                          // count=1, user mode
  g0[1] = 0u;                                          // lds_addr = 0
  g0[2] = (unsigned int)(ga & 0xFFFFFFFFull);
  g0[3] = (unsigned int)((ga >> 32) & 0x1FFFFFFull) | (2u << 30);  // type=2
  v8i g1;
  g1[0] = (1 << 16)      // data_size = 2 bytes
        | (1 << 20)      // pad_enable
        | (5 << 22)      // pad_interval: every 64 DWORDs
        | (3 << 25);     // pad_amount: 4 DWORDs
  g1[1] = (K & 0xFFFF) << 16;                          // tensor_dim0 lo
  g1[2] = ((unsigned)K >> 16) | ((nrows & 0xFFFF) << 16); // dim0 hi | dim1 lo
  g1[3] = ((unsigned)nrows >> 16) | (128 << 16);       // dim1 hi | tile_dim0=128
  g1[4] = nrows;                                       // tile_dim1
  g1[5] = K;                                           // tensor_dim0_stride lo
  g1[6] = 0;
  g1[7] = 0;
  v4i z = {0, 0, 0, 0};
#if __clang_major__ >= 23
  v8i z8 = {0, 0, 0, 0, 0, 0, 0, 0};
  __builtin_amdgcn_tensor_load_to_lds(g0, g1, z, z, z8, 0);
#else
  __builtin_amdgcn_tensor_load_to_lds(g0, g1, z, z, 0);
#endif
}

// ---------------------------------------------------------------------------
// Pack fp32 W[K][N] (row-major) -> bf16 W^T[N][K]
// ---------------------------------------------------------------------------
__global__ void __launch_bounds__(256) pack_wt_kernel(
    const float* __restrict__ W, __bf16* __restrict__ WT, int K, int N) {
  const int idx = blockIdx.x * 256 + threadIdx.x;
  if (idx >= K * N) return;
  const int n = idx / K;
  const int k = idx - n * K;
  WT[idx] = (__bf16)W[(size_t)k * (size_t)N + n];
}

// ---------------------------------------------------------------------------
// C[M,N] = act(A[M,K] @ W[K,N] + bias[N]); WT = bf16 [N][K]; act 1 = relu.
// grid.x = M/16; block (32, N/16). Weights staged via TDM, 128 K-cols at a time.
// ---------------------------------------------------------------------------
__global__ void __launch_bounds__(256) gemm_bias_act_kernel(
    const float* __restrict__ A, const __bf16* __restrict__ WT,
    const float* __restrict__ bias, float* __restrict__ C,
    int K, int N, int act) {
  __shared__ __align__(16) __bf16 sW[128 * SW_STRIDE];
  const int m0   = blockIdx.x * 16;
  const int n0   = threadIdx.y * 16;
  const int lane = threadIdx.x;
  v8f acc = {};
  for (int kc = 0; kc < K; kc += 128) {
    if (kc) __syncthreads();                 // protect LDS before restage
    if (threadIdx.y == 0) {
      tdm_stage_wt(WT + kc, K, N);           // tile start at column kc of W^T
      __builtin_amdgcn_s_wait_tensorcnt(0);
    }
    __syncthreads();
#pragma unroll
    for (int kk = 0; kk < 128; kk += 32) {
      v16bf a = load_frag_A(A, K, m0, kc + kk);
      v16bf b = load_frag_B_lds(sW, kk, n0);
      acc = __builtin_amdgcn_wmma_f32_16x16x32_bf16(false, a, false, b,
                                                    (short)0, acc, false, false);
    }
  }
  const int   n  = n0 + (lane & 15);
  const float bv = bias[n];
#pragma unroll
  for (int v = 0; v < 8; ++v) {
    const int m = m0 + v + 8 * (lane >> 4);
    float x = acc[v] + bv;
    if (act) x = fmaxf(x, 0.f);
    C[(size_t)m * (size_t)N + n] = x;
  }
}

// ---------------------------------------------------------------------------
// rel[n,0,:] = d0[n]*x[n,:]; rel[n,1,:] = d1[n]*x[n,:]; also zero score acc.
// ---------------------------------------------------------------------------
__global__ void __launch_bounds__(256) init_rel_kernel(
    const float* __restrict__ x, const float* __restrict__ d0,
    const float* __restrict__ d1, float* __restrict__ rel,
    int Nn, float* __restrict__ s_sum) {
  if (blockIdx.x == 0 && threadIdx.x == 0) { s_sum[0] = 0.f; s_sum[1] = 0.f; }
  const int idx = blockIdx.x * blockDim.x + threadIdx.x;
  if (idx >= Nn * 32) return;
  const int n = idx >> 5;
  const int c = idx & 31;
  const float4 xv = ((const float4*)x)[(size_t)n * 32 + c];
  const float a = d0[n], b = d1[n];
  float4* out = (float4*)rel;
  out[(size_t)n * 64 + c]      = make_float4(a * xv.x, a * xv.y, a * xv.z, a * xv.w);
  out[(size_t)n * 64 + 32 + c] = make_float4(b * xv.x, b * xv.y, b * xv.z, b * xv.w);
}

// ---------------------------------------------------------------------------
// rel[dst[e], r, :] += w[e] * h[src[e], :]   (global_atomic_add_f32 scatter)
// ---------------------------------------------------------------------------
__global__ void __launch_bounds__(256) spmm_scatter_kernel(
    const int* __restrict__ src, const int* __restrict__ dst,
    const float* __restrict__ w, const float* __restrict__ h,
    float* __restrict__ rel, int r, int E) {
  const int e = blockIdx.x * blockDim.y + threadIdx.y;
  if (e >= E) return;
  const int   s  = src[e];
  const int   d  = dst[e];
  const float wv = w[e];
  const float4 v = ((const float4*)(h + (size_t)s * 128))[threadIdx.x];
  float* o = rel + (size_t)d * 256 + r * 128 + threadIdx.x * 4;
  atomicAdd(o + 0, wv * v.x);
  atomicAdd(o + 1, wv * v.y);
  atomicAdd(o + 2, wv * v.z);
  atomicAdd(o + 3, wv * v.w);
}

// ---------------------------------------------------------------------------
// s_sum[r] += sum over 16-row tile of tanh(rel[n,r,:]@W1 + b1) . w2
// W1T = bf16 [128][128]; rel viewed as A with lda=256, slice offset r*128.
// ---------------------------------------------------------------------------
__global__ void __launch_bounds__(256) sem_score_kernel(
    const float* __restrict__ rel, int r, const __bf16* __restrict__ W1T,
    const float* __restrict__ b1, const float* __restrict__ w2,
    float* __restrict__ s_sum) {
  __shared__ __align__(16) unsigned char smem[128 * SW_STRIDE * 2 + 64];
  __bf16* sW     = (__bf16*)smem;
  float*  rowsum = (float*)(smem + 128 * SW_STRIDE * 2);
  const int m0   = blockIdx.x * 16;
  const int n0   = threadIdx.y * 16;
  const int lane = threadIdx.x;
  const int tid  = threadIdx.y * 32 + lane;
  if (tid < 16) rowsum[tid] = 0.f;
  if (threadIdx.y == 0) {
    tdm_stage_wt(W1T, 128, 128);
    __builtin_amdgcn_s_wait_tensorcnt(0);
  }
  __syncthreads();
  const float* A = rel + r * 128;
  v8f acc = {};
#pragma unroll
  for (int kk = 0; kk < 128; kk += 32) {
    v16bf a = load_frag_A(A, 256, m0, kk);
    v16bf b = load_frag_B_lds(sW, kk, n0);
    acc = __builtin_amdgcn_wmma_f32_16x16x32_bf16(false, a, false, b,
                                                  (short)0, acc, false, false);
  }
  const int   n  = n0 + (lane & 15);
  const float bv = b1[n];
  const float wv = w2[n];
#pragma unroll
  for (int v = 0; v < 8; ++v) {
    const int mrow = v + 8 * (lane >> 4);
    atomicAdd(&rowsum[mrow], tanhf(acc[v] + bv) * wv);
  }
  __syncthreads();
  if (tid == 0) {
    float t = 0.f;
#pragma unroll
    for (int i = 0; i < 16; ++i) t += rowsum[i];
    atomicAdd(&s_sum[r], t);
  }
}

// beta = softmax(s_sum / Nn) over the 2 relations
__global__ void beta_kernel(const float* __restrict__ s_sum,
                            float* __restrict__ beta, float invN) {
  if (threadIdx.x == 0) {
    const float a = s_sum[0] * invN, b = s_sum[1] * invN;
    const float mx = fmaxf(a, b);
    const float ea = expf(a - mx), eb = expf(b - mx);
    const float inv = 1.f / (ea + eb);
    beta[0] = ea * inv;
    beta[1] = eb * inv;
  }
}

// h[n,:] = beta[0]*rel[n,0,:] + beta[1]*rel[n,1,:]
__global__ void __launch_bounds__(256) combine_kernel(
    const float* __restrict__ rel, const float* __restrict__ beta,
    float* __restrict__ h, int Nn) {
  const int idx = blockIdx.x * blockDim.x + threadIdx.x;
  if (idx >= Nn * 32) return;
  const int n = idx >> 5;
  const int c = idx & 31;
  const float b0 = beta[0], b1 = beta[1];
  const float4* r4 = (const float4*)rel;
  const float4 v0 = r4[(size_t)n * 64 + c];
  const float4 v1 = r4[(size_t)n * 64 + 32 + c];
  ((float4*)h)[(size_t)n * 32 + c] =
      make_float4(b0 * v0.x + b1 * v1.x, b0 * v0.y + b1 * v1.y,
                  b0 * v0.z + b1 * v1.z, b0 * v0.w + b1 * v1.w);
}

// ---------------------------------------------------------------------------
extern "C" void kernel_launch(void* const* d_in, const int* in_sizes, int n_in,
                              void* d_out, int out_size, void* d_ws, size_t ws_size,
                              hipStream_t stream) {
  const float* x_paper = (const float*)d_in[0];
  const float* x_author= (const float*)d_in[1];
  const float* fc1p_W  = (const float*)d_in[2];
  const float* fc1p_b  = (const float*)d_in[3];
  const float* fc1a_W  = (const float*)d_in[4];
  const float* fc1a_b  = (const float*)d_in[5];
  const float* fc2_W   = (const float*)d_in[6];
  const float* fc2_b   = (const float*)d_in[7];
  const float* semW1   = (const float*)d_in[8];
  const float* semb1   = (const float*)d_in[9];
  const float* semw2   = (const float*)d_in[10];
  const float* d_pp    = (const float*)d_in[11];
  const float* d_pa    = (const float*)d_in[12];
  const float* d_ap    = (const float*)d_in[13];
  const float* d_aa    = (const float*)d_in[14];
  const float* w_pp    = (const float*)d_in[15];
  const float* w_pa    = (const float*)d_in[16];
  const float* w_ap    = (const float*)d_in[17];
  const float* w_aa    = (const float*)d_in[18];
  const int* src_pp = (const int*)d_in[19];
  const int* dst_pp = (const int*)d_in[20];
  const int* src_pa = (const int*)d_in[21];
  const int* dst_pa = (const int*)d_in[22];
  const int* src_ap = (const int*)d_in[23];
  const int* dst_ap = (const int*)d_in[24];
  const int* src_aa = (const int*)d_in[25];
  const int* dst_aa = (const int*)d_in[26];

  const int NP  = in_sizes[0] / 256;          // 100000
  const int NA  = in_sizes[1] / 128;          // 100000
  const int E   = in_sizes[15];               // 1600000
  const int HOP = in_sizes[9] / (2 * 128);    // 5
  const int H   = 128;

  // Workspace layout: fp32 buffers, then bf16 packed-transposed weights.
  float* ws     = (float*)d_ws;
  float* x_p    = ws;                               // NP*H
  float* x_a    = x_p    + (size_t)NP * H;          // NA*H
  float* h_p    = x_a    + (size_t)NA * H;          // NP*H
  float* h_a    = h_p    + (size_t)NP * H;          // NA*H
  float* hp_rel = h_a    + (size_t)NA * H;          // NP*2*H
  float* ha_rel = hp_rel + (size_t)NP * 2 * H;      // NA*2*H
  float* s_sum  = ha_rel + (size_t)NA * 2 * H;      // 2
  float* beta   = s_sum + 2;                        // 2
  __bf16* fc1p_T = (__bf16*)(beta + 2);             // 128*256 (16B-aligned)
  __bf16* fc1a_T = fc1p_T + 128 * 256;              // 128*128
  __bf16* fc2_T  = fc1a_T + 128 * 128;              // 64*128
  __bf16* sem_T  = fc2_T  + 64 * 128;               // HOP*2 * 128*128

  const dim3 wmma8(32, 8);   // 8 waves -> 128 output columns
  const dim3 wmma4(32, 4);   // 4 waves -> 64 output columns
  const int  ev_grid = (NP * 32 + 255) / 256;
  const int  sp_grid = (E + 7) / 8;

  // One-time (per launch) weight packing: fp32 [K][N] -> bf16 [N][K]
  pack_wt_kernel<<<(256 * 128 + 255) / 256, 256, 0, stream>>>(fc1p_W, fc1p_T, 256, 128);
  pack_wt_kernel<<<(128 * 128 + 255) / 256, 256, 0, stream>>>(fc1a_W, fc1a_T, 128, 128);
  pack_wt_kernel<<<(128 * 64  + 255) / 256, 256, 0, stream>>>(fc2_W,  fc2_T,  128, 64);
  for (int m = 0; m < HOP * 2; ++m)
    pack_wt_kernel<<<(128 * 128 + 255) / 256, 256, 0, stream>>>(
        semW1 + (size_t)m * 128 * 128, sem_T + (size_t)m * 128 * 128, 128, 128);

  // Input projections
  gemm_bias_act_kernel<<<NP / 16, wmma8, 0, stream>>>(x_paper,  fc1p_T, fc1p_b, x_p, 256, H, 1);
  gemm_bias_act_kernel<<<NA / 16, wmma8, 0, stream>>>(x_author, fc1a_T, fc1a_b, x_a, 128, H, 1);
  hipMemcpyAsync(h_p, x_p, (size_t)NP * H * sizeof(float), hipMemcpyDeviceToDevice, stream);
  hipMemcpyAsync(h_a, x_a, (size_t)NA * H * sizeof(float), hipMemcpyDeviceToDevice, stream);

  for (int i = 0; i < HOP; ++i) {
    const __bf16* W1pT = sem_T + (size_t)(i * 2 + 0) * 128 * 128;
    const __bf16* W1aT = sem_T + (size_t)(i * 2 + 1) * 128 * 128;
    const float* b1p = semb1 + (size_t)(i * 2 + 0) * 128;
    const float* b1a = semb1 + (size_t)(i * 2 + 1) * 128;
    const float* w2p = semw2 + (size_t)(i * 2 + 0) * 128;
    const float* w2a = semw2 + (size_t)(i * 2 + 1) * 128;

    // paper node type: relations (p<-p via h_p, p<-a via h_a)
    init_rel_kernel<<<ev_grid, 256, 0, stream>>>(x_p, d_pp, d_pa, hp_rel, NP, s_sum);
    spmm_scatter_kernel<<<sp_grid, wmma8, 0, stream>>>(src_pp, dst_pp, w_pp, h_p, hp_rel, 0, E);
    spmm_scatter_kernel<<<sp_grid, wmma8, 0, stream>>>(src_pa, dst_pa, w_pa, h_a, hp_rel, 1, E);
    sem_score_kernel<<<NP / 16, wmma8, 0, stream>>>(hp_rel, 0, W1pT, b1p, w2p, s_sum);
    sem_score_kernel<<<NP / 16, wmma8, 0, stream>>>(hp_rel, 1, W1pT, b1p, w2p, s_sum);
    beta_kernel<<<1, 32, 0, stream>>>(s_sum, beta, 1.0f / (float)NP);
    combine_kernel<<<ev_grid, 256, 0, stream>>>(hp_rel, beta, h_p, NP);

    // author node type: relations (a<-p via UPDATED h_p, a<-a via h_a)
    init_rel_kernel<<<ev_grid, 256, 0, stream>>>(x_a, d_ap, d_aa, ha_rel, NA, s_sum);
    spmm_scatter_kernel<<<sp_grid, wmma8, 0, stream>>>(src_ap, dst_ap, w_ap, h_p, ha_rel, 0, E);
    spmm_scatter_kernel<<<sp_grid, wmma8, 0, stream>>>(src_aa, dst_aa, w_aa, h_a, ha_rel, 1, E);
    sem_score_kernel<<<NA / 16, wmma8, 0, stream>>>(ha_rel, 0, W1aT, b1a, w2a, s_sum);
    sem_score_kernel<<<NA / 16, wmma8, 0, stream>>>(ha_rel, 1, W1aT, b1a, w2a, s_sum);
    beta_kernel<<<1, 32, 0, stream>>>(s_sum, beta, 1.0f / (float)NA);
    combine_kernel<<<ev_grid, 256, 0, stream>>>(ha_rel, beta, h_a, NA);
  }

  // Final: out = h_p @ fc2_W + fc2_b   [NP, 64]
  gemm_bias_act_kernel<<<NP / 16, wmma4, 0, stream>>>(h_p, fc2_T, fc2_b, (float*)d_out, 128, 64, 0);
}
